// EmbeddingLayer_10788957847952
// MI455X (gfx1250) — matrix operand order repaired and checked by reading, not compiled, not run on hardware
//
#include <hip/hip_runtime.h>
#include <cstdint>

#define Bsz 128
#define Tsz 512
#define Esz 512
#define OUTsz 512
#define E3 1536
#define E2 1024

typedef unsigned short u16;
typedef __attribute__((ext_vector_type(16))) __bf16 v16bf;
typedef __attribute__((ext_vector_type(8)))  __bf16 v8bf;
typedef __attribute__((ext_vector_type(8)))  float  v8f;

// ---------- helpers ----------

__device__ __forceinline__ u16 f32_to_bf16(float f) {
    unsigned int u = __float_as_uint(f);
    unsigned int lsb = (u >> 16) & 1u;
    u += 0x7fffu + lsb;               // round-to-nearest-even
    return (u16)(u >> 16);
}

// A fragment: 16x32 bf16, row-major source, lda = K (elements).
// lanes 0-15: row M=lane, K = 0..7 (v0-3) and 16..23 (v4-7)
// lanes 16-31: row M=lane-16, K = 8..15 and 24..31
__device__ __forceinline__ v16bf ld_frag_a(const u16* p) {
    union { v16bf v; v8bf h[2]; } u;
    u.h[0] = *(const v8bf*)(p);
    u.h[1] = *(const v8bf*)(p + 16);
    return u.v;
}

// B fragment: 32x16 bf16 from transposed weights Bt[N][K] (row-major, ld = K).
// lanes 0-15: column N=lane, K = k..k+7 and k+16..k+23; lanes 16-31: col N=lane-16, K = k+8..k+15 and k+24..k+31
__device__ __forceinline__ v16bf ld_frag_b(const u16* p) {
    union { v16bf v; v8bf h[2]; } u;
    u.h[0] = *(const v8bf*)(p);
    u.h[1] = *(const v8bf*)(p + 16);
    return u.v;
}

__device__ __forceinline__ float sigmoidf_(float x) {
    return 1.0f / (1.0f + __expf(-x));
}

// ---------- kernels ----------

__global__ void embed_bf16_kernel(const int* __restrict__ tok,
                                  const float* __restrict__ emb,
                                  u16* __restrict__ xbf, int total) {
    int i = blockIdx.x * blockDim.x + threadIdx.x;
    if (i >= total) return;
    int row = i >> 9;            // /E (E=512)
    int e   = i & (Esz - 1);
    xbf[i] = f32_to_bf16(emb[(long)tok[row] * Esz + e]);
}

// out[n*K + k] = bf16(in[k*N + n])   (weight transpose + convert)
__global__ void transpose_bf16_kernel(const float* __restrict__ in,
                                      u16* __restrict__ out, int N_, int K_) {
    int i = blockIdx.x * blockDim.x + threadIdx.x;
    if (i >= N_ * K_) return;
    int n = i / K_;
    int k = i - n * K_;
    out[i] = f32_to_bf16(in[k * N_ + n]);
}

__global__ void zero_h_kernel(float* __restrict__ hf, u16* __restrict__ hb, int count) {
    int i = blockIdx.x * blockDim.x + threadIdx.x;
    if (i >= count) return;
    hf[i] = 0.0f;
    hb[i] = 0;
}

// C[M,N] (f32) = A_bf[M,K] @ Bt_bf[N,K]^T + bias[N]
// wave tile = 32(M) x 64(N): 2 A-frags x 4 B-frags -> 8 WMMA per K-chunk.
// blockDim = (32,4): block tile 128(M) x 64(N).
__global__ void gemm_bf16_wmma_kernel(const u16* __restrict__ A,
                                      const u16* __restrict__ Bt,
                                      const float* __restrict__ bias,
                                      float* __restrict__ C,
                                      int M, int N, int K) {
    const int lane = threadIdx.x & 31;
    const int m0 = (blockIdx.y * blockDim.y + threadIdx.y) * 32;
    const int n0 = blockIdx.x * 64;
    if (m0 >= M) return;

    const int mA  = lane & 15;
    const int kgA = (lane >> 4) << 3;   // 0 or 8
    const int nB  = lane & 15;
    const int khB = (lane >> 4) << 3;   // 0 or 8

    const u16* Ap0 = A + (size_t)(m0 +  0 + mA) * K + kgA;
    const u16* Ap1 = A + (size_t)(m0 + 16 + mA) * K + kgA;
    const u16* Bp0 = Bt + (size_t)(n0 +  0 + nB) * K + khB;
    const u16* Bp1 = Bt + (size_t)(n0 + 16 + nB) * K + khB;
    const u16* Bp2 = Bt + (size_t)(n0 + 32 + nB) * K + khB;
    const u16* Bp3 = Bt + (size_t)(n0 + 48 + nB) * K + khB;

    v8f acc[2][4] = {};
    for (int kk = 0; kk < K; kk += 32) {
        v16bf a0 = ld_frag_a(Ap0 + kk);
        v16bf a1 = ld_frag_a(Ap1 + kk);
        v16bf b0 = ld_frag_b(Bp0 + kk);
        v16bf b1 = ld_frag_b(Bp1 + kk);
        v16bf b2 = ld_frag_b(Bp2 + kk);
        v16bf b3 = ld_frag_b(Bp3 + kk);
        acc[0][0] = __builtin_amdgcn_wmma_f32_16x16x32_bf16(false, a0, false, b0, (short)0, acc[0][0], false, false);
        acc[0][1] = __builtin_amdgcn_wmma_f32_16x16x32_bf16(false, a0, false, b1, (short)0, acc[0][1], false, false);
        acc[0][2] = __builtin_amdgcn_wmma_f32_16x16x32_bf16(false, a0, false, b2, (short)0, acc[0][2], false, false);
        acc[0][3] = __builtin_amdgcn_wmma_f32_16x16x32_bf16(false, a0, false, b3, (short)0, acc[0][3], false, false);
        acc[1][0] = __builtin_amdgcn_wmma_f32_16x16x32_bf16(false, a1, false, b0, (short)0, acc[1][0], false, false);
        acc[1][1] = __builtin_amdgcn_wmma_f32_16x16x32_bf16(false, a1, false, b1, (short)0, acc[1][1], false, false);
        acc[1][2] = __builtin_amdgcn_wmma_f32_16x16x32_bf16(false, a1, false, b2, (short)0, acc[1][2], false, false);
        acc[1][3] = __builtin_amdgcn_wmma_f32_16x16x32_bf16(false, a1, false, b3, (short)0, acc[1][3], false, false);
    }

    const int cn = lane & 15;
    const int msub = (lane >> 4) << 3;
#pragma unroll
    for (int nt = 0; nt < 4; ++nt) {
        int col = n0 + nt * 16 + cn;
        float bv = bias[col];
#pragma unroll
        for (int mt = 0; mt < 2; ++mt) {
            int mrow = m0 + mt * 16 + msub;
#pragma unroll
            for (int i = 0; i < 8; ++i) {
                C[(size_t)(mrow + i) * N + col] = acc[mt][nt][i] + bv;
            }
        }
    }
}

// One GRU timestep, both directions fused (blockIdx.z = dir).
// grid = (E/16, B/16, 2), block = 32 (one wave).
__global__ void gru_step_kernel(const u16* __restrict__ Ut_f, const u16* __restrict__ Ut_b,
                                const float* __restrict__ bU_f, const float* __restrict__ bU_b,
                                const float* __restrict__ xpf, const float* __restrict__ xpb,
                                float* __restrict__ hf32, u16* __restrict__ hbf,
                                u16* __restrict__ ycat, int t) {
    const int dir = blockIdx.z;
    const u16*   Ut = dir ? Ut_b : Ut_f;
    const float* bU = dir ? bU_b : bU_f;
    const float* xp = dir ? xpb  : xpf;
    const int t_x = dir ? (Tsz - 1 - t) : t;

    const int lane = threadIdx.x & 31;
    const int m0 = blockIdx.y * 16;   // batch tile
    const int n0 = blockIdx.x * 16;   // hidden-unit tile

    const int mA  = lane & 15;
    const int kgA = (lane >> 4) << 3;
    const int nB  = lane & 15;
    const int khB = (lane >> 4) << 3;

    const u16* Ap = hbf + (size_t)dir * Bsz * Esz + (size_t)(m0 + mA) * Esz + kgA;
    const u16* Bz = Ut + (size_t)(0 * Esz + n0 + nB) * Esz + khB;
    const u16* Br = Ut + (size_t)(1 * Esz + n0 + nB) * Esz + khB;
    const u16* Bh = Ut + (size_t)(2 * Esz + n0 + nB) * Esz + khB;

    v8f az = {}, ar = {}, ah = {};
    for (int kk = 0; kk < Esz; kk += 32) {
        v16bf a  = ld_frag_a(Ap + kk);
        v16bf bz = ld_frag_b(Bz + kk);
        v16bf br = ld_frag_b(Br + kk);
        v16bf bh = ld_frag_b(Bh + kk);
        az = __builtin_amdgcn_wmma_f32_16x16x32_bf16(false, a, false, bz, (short)0, az, false, false);
        ar = __builtin_amdgcn_wmma_f32_16x16x32_bf16(false, a, false, br, (short)0, ar, false, false);
        ah = __builtin_amdgcn_wmma_f32_16x16x32_bf16(false, a, false, bh, (short)0, ah, false, false);
    }

    const int n   = n0 + (lane & 15);
    const int mrb = m0 + ((lane >> 4) << 3);
    const float buz = bU[n];
    const float bur = bU[Esz + n];
    const float buh = bU[2 * Esz + n];

#pragma unroll
    for (int i = 0; i < 8; ++i) {
        int b = mrb + i;
        size_t xbase = ((size_t)b * Tsz + t_x) * E3;
        float xz = xp[xbase + n];
        float xr = xp[xbase + Esz + n];
        float xh = xp[xbase + 2 * Esz + n];

        float z = sigmoidf_(xz + az[i] + buz);
        float r = sigmoidf_(xr + ar[i] + bur);
        float hh = tanhf(xh + r * (ah[i] + buh));

        size_t hidx = (size_t)dir * Bsz * Esz + (size_t)b * Esz + n;
        float hold = hf32[hidx];
        float hnew = z * hold + (1.0f - z) * hh;
        hf32[hidx] = hnew;
        u16 hb = f32_to_bf16(hnew);
        hbf[hidx] = hb;
        ycat[((size_t)b * Tsz + t_x) * E2 + (size_t)dir * Esz + n] = hb;
    }
}

// ---------- launcher ----------

extern "C" void kernel_launch(void* const* d_in, const int* in_sizes, int n_in,
                              void* d_out, int out_size, void* d_ws, size_t ws_size,
                              hipStream_t stream) {
    (void)in_sizes; (void)n_in; (void)out_size; (void)ws_size;

    const int*   tok = (const int*)  d_in[0];
    const float* emb = (const float*)d_in[1];
    const float* Wf  = (const float*)d_in[2];
    const float* Uf  = (const float*)d_in[3];
    const float* bWf = (const float*)d_in[4];
    const float* bUf = (const float*)d_in[5];
    const float* Wb  = (const float*)d_in[6];
    const float* Ub  = (const float*)d_in[7];
    const float* bWb = (const float*)d_in[8];
    const float* bUb = (const float*)d_in[9];
    const float* Wd  = (const float*)d_in[10];
    const float* bd  = (const float*)d_in[11];
    float* out = (float*)d_out;

    // workspace carve-up (256B aligned slices)
    char* w = (char*)d_ws;
    size_t off = 0;
    auto carve = [&](size_t bytes) -> char* {
        char* p = w + off;
        off += (bytes + 255) & ~(size_t)255;
        return p;
    };
    const size_t BT = (size_t)Bsz * Tsz;
    u16*   x_bf  = (u16*)  carve(BT * Esz * 2);          // [B*T, E] bf16
    u16*   Wf_t  = (u16*)  carve((size_t)E3 * Esz * 2);  // [3E, E] bf16
    u16*   Uf_t  = (u16*)  carve((size_t)E3 * Esz * 2);
    u16*   Wb_t  = (u16*)  carve((size_t)E3 * Esz * 2);
    u16*   Ub_t  = (u16*)  carve((size_t)E3 * Esz * 2);
    u16*   Wd_t  = (u16*)  carve((size_t)OUTsz * E2 * 2); // [OUT, 2E] bf16
    float* xpf   = (float*)carve(BT * E3 * 4);           // [B*T, 3E] f32
    float* xpb   = (float*)carve(BT * E3 * 4);
    u16*   ycat  = (u16*)  carve(BT * E2 * 2);           // [B*T, 2E] bf16
    float* h_f32 = (float*)carve((size_t)2 * Bsz * Esz * 4);
    u16*   h_bf  = (u16*)  carve((size_t)2 * Bsz * Esz * 2);

    // 1) embedding gather -> bf16
    {
        int total = (int)(BT * Esz);
        embed_bf16_kernel<<<(total + 255) / 256, 256, 0, stream>>>(tok, emb, x_bf, total);
    }
    // 2) transpose+convert weights
    {
        int tW = E3 * Esz;
        transpose_bf16_kernel<<<(tW + 255) / 256, 256, 0, stream>>>(Wf, Wf_t, E3, Esz);
        transpose_bf16_kernel<<<(tW + 255) / 256, 256, 0, stream>>>(Uf, Uf_t, E3, Esz);
        transpose_bf16_kernel<<<(tW + 255) / 256, 256, 0, stream>>>(Wb, Wb_t, E3, Esz);
        transpose_bf16_kernel<<<(tW + 255) / 256, 256, 0, stream>>>(Ub, Ub_t, E3, Esz);
        int tD = OUTsz * E2;
        transpose_bf16_kernel<<<(tD + 255) / 256, 256, 0, stream>>>(Wd, Wd_t, OUTsz, E2);
    }
    // 3) input projections xp = x @ W + bW for both directions
    {
        dim3 blk(32, 4);
        dim3 grd(E3 / 64, (unsigned)(BT / 128));
        gemm_bf16_wmma_kernel<<<grd, blk, 0, stream>>>(x_bf, Wf_t, bWf, xpf, (int)BT, E3, Esz);
        gemm_bf16_wmma_kernel<<<grd, blk, 0, stream>>>(x_bf, Wb_t, bWb, xpb, (int)BT, E3, Esz);
    }
    // 4) init hidden state
    {
        int cnt = 2 * Bsz * Esz;
        zero_h_kernel<<<(cnt + 255) / 256, 256, 0, stream>>>(h_f32, h_bf, cnt);
    }
    // 5) sequential recurrence, both directions per step
    {
        dim3 grd(Esz / 16, Bsz / 16, 2);
        for (int t = 0; t < Tsz; ++t) {
            gru_step_kernel<<<grd, 32, 0, stream>>>(Uf_t, Ub_t, bUf, bUb,
                                                    xpf, xpb, h_f32, h_bf, ycat, t);
        }
    }
    // 6) output dense: out = ycat @ Wd + bd
    {
        dim3 blk(32, 4);
        dim3 grd(OUTsz / 64, (unsigned)(BT / 128));
        gemm_bf16_wmma_kernel<<<grd, blk, 0, stream>>>(ycat, Wd_t, bd, out, (int)BT, OUTsz, E2);
    }
}
